// LCNNRegressor_88622355186235
// MI455X (gfx1250) — matrix operand order repaired and checked by reading, not compiled, not run on hardware
//
#include <hip/hip_runtime.h>
#include <hip/hip_bf16.h>
#include <hip/hip_fp16.h>

// ---------------------------------------------------------------------------
// LCNN phase regressor, fully fused for gfx1250 (MI455X).
//  - wave32; conv2 + bilinear contractions on v_wmma_f32_16x16x32_f16
//  - one workgroup = one 32x32 output tile of one (b,s) slice; conv1 computed
//    on a 34x34 halo in LDS so NO intermediate tensors ever touch HBM
//    (traffic = 17 MB input reads + 32 KB of pooled atomics, vs ~1 GB if the
//    angle maps were materialized; 23.3 TB/s HBM makes this compute-bound).
//  - 32x32 tiles amortize the halo: 13% redundant conv1 work vs 27% at 16x16.
//  - A-matrix WMMA fragments are pre-swizzled by a setup kernel into a contiguous
//    per-lane layout (16-bit A 16x32: lane<16 holds K=0..15 of row
//    M=lane, lane>=16 holds K=16..31 of row M=lane-16; B 32x16:
//    lane<16 holds K=0..15 of column N=lane, lane>=16 holds K=16..31 of
//    column N=lane-16), so every fragment load is a contiguous 32 B b128x2.
// ---------------------------------------------------------------------------

typedef __attribute__((ext_vector_type(16))) _Float16 v16h;
typedef __attribute__((ext_vector_type(8)))  _Float16 v8h;
typedef __attribute__((ext_vector_type(8)))  float    v8f;

#define LSZ   256      // image side
#define TILE  32       // output tile side per workgroup
#define H1    34       // TILE + 2 (conv2 halo)
#define H2    36       // TILE + 4 (conv1+conv2 halo)

// d_ws layout
#define WS_POOLED_OFF 0        // 256 floats  (32 b x 8 ch partial sums)
#define WS_AFRAG2_OFF 1024     // 5*32*16 f16 (conv2 A fragments, tap pairs)
#define WS_AFRAGB_OFF 6144     // 8*2*32*16 f16 (bilinear A fragments u/v)

// K mapping of the 16-bit A fragment (contiguous, matching B): half j of lane L maps to
//   K = j + (L>=16 ? 16 : 0),  row M = L & 15.
__device__ __host__ inline int afrag_k(int j, int lane) {
    return j + ((lane >= 16) ? 16 : 0);
}

// ---------------------------------------------------------------------------
// Kernel 0: build WMMA A-fragments in d_ws, zero the pooled accumulators.
// ---------------------------------------------------------------------------
__global__ void lcnn_setup_kernel(const float* __restrict__ w2,   // (16,16,3,3)
                                  const float* __restrict__ wb,   // (8,16,16)
                                  float* __restrict__ pooled,
                                  _Float16* __restrict__ afrag2,
                                  _Float16* __restrict__ afragb) {
    const int tid = threadIdx.x;
    pooled[tid] = 0.0f;   // 256 entries, 256 threads

    // conv2 fragments: K tile packs a pair of 3x3 taps: K<16 -> (tap t0, c=K),
    // K>=16 -> (tap t1, c=K-16). tp=4 has only tap 8; its upper half is zero.
    for (int e = tid; e < 5 * 32; e += 256) {
        const int tp = e >> 5, lane = e & 31;
        const int o = lane & 15;
        const int t0 = 2 * tp;
        const int t1 = (2 * tp + 1 < 9) ? (2 * tp + 1) : -1;
        _Float16* dst = afrag2 + e * 16;
        for (int j = 0; j < 16; ++j) {
            const int K   = afrag_k(j, lane);
            const int tap = (K < 16) ? t0 : t1;
            const int c   = K & 15;
            float val = 0.0f;
            if (tap >= 0) val = w2[(o * 16 + c) * 9 + tap];
            dst[j] = (_Float16)val;
        }
    }

    // bilinear fragments: A_u = [W_o | 0], A_v = [0 | W_o] against the packed
    // B fragment (K<16 = u2[c], K>=16 = v2[c]).
    for (int e = tid; e < 8 * 2 * 32; e += 256) {
        const int o = e >> 6, uv = (e >> 5) & 1, lane = e & 31;
        const int cp = lane & 15;
        _Float16* dst = afragb + e * 16;
        for (int j = 0; j < 16; ++j) {
            const int K = afrag_k(j, lane);
            float val = 0.0f;
            if (uv == 0 && K < 16)  val = wb[(o * 16 + cp) * 16 + K];
            if (uv == 1 && K >= 16) val = wb[(o * 16 + cp) * 16 + (K - 16)];
            dst[j] = (_Float16)val;
        }
    }
}

// ---------------------------------------------------------------------------
// Kernel 1: fused conv1 -> conv2(WMMA) -> bilinear(WMMA) -> pooled atomics.
// Grid: 64 slices * 64 tiles = 4096 blocks of 256 threads (8 waves).
// Each wave owns 8 row-strips of 16 pixels (64 strips per 32x32 tile).
// ---------------------------------------------------------------------------
__launch_bounds__(256)
__global__ void lcnn_fused_kernel(const float* __restrict__ x,
                                  const float* __restrict__ w1,  // (16,1,3,3)
                                  const float* __restrict__ b1,  // (16)
                                  const float* __restrict__ b2,  // (16)
                                  const float* __restrict__ b3,  // (8)
                                  const _Float16* __restrict__ afrag2,
                                  const _Float16* __restrict__ afragb,
                                  float* __restrict__ pooled) {
    __shared__ float cs[H2][H2], sn[H2][H2];        // cos/sin(x), zero-padded
    __shared__ float w1s[16][9];
    __shared__ float cb1s[16], sb1s[16], cb2s[16], sb2s[16], bb3s[8];
    __shared__ __align__(32) _Float16 U1[H1 * H1 * 16];  // [y][x][c] pixel-major
    __shared__ __align__(32) _Float16 V1[H1 * H1 * 16];
    __shared__ __align__(32) _Float16 ABs[8 * 2 * 32 * 16]; // bilinear A frags
    __shared__ __align__(32) _Float16 SU[8][16][16];  // per-wave u2 staging
    __shared__ __align__(32) _Float16 SV[8][16][16];  // per-wave v2 staging
    __shared__ float red[8][8];

    const int tid  = threadIdx.x;
    const int lane = tid & 31;
    const int wave = tid >> 5;
    const int p    = lane & 15;   // WMMA column = pixel within row strip
    const int hi   = lane >> 4;   // lane-half selector

    const int slice = blockIdx.x >> 6;              // = b*2 + s   (64 tiles/slice)
    const int tidx  = blockIdx.x & 63;
    const int ty    = (tidx >> 3) << 5;             // 8x8 grid of 32x32 tiles
    const int tx    = (tidx & 7) << 5;
    const float* xs = x + (size_t)slice * (LSZ * LSZ);

    // ---- constants ----
    if (tid < 16) {
        cb1s[tid] = __cosf(b1[tid]); sb1s[tid] = __sinf(b1[tid]);
        cb2s[tid] = __cosf(b2[tid]); sb2s[tid] = __sinf(b2[tid]);
    }
    if (tid < 8)   bb3s[tid] = b3[tid];
    if (tid < 144) w1s[tid / 9][tid % 9] = w1[tid];

    // stage bilinear A fragments into LDS (16 KB, b128 copies)
    {
        const uint4* src = (const uint4*)afragb;
        uint4*       dst = (uint4*)ABs;
        for (int i = tid; i < (8 * 2 * 32 * 16 * 2) / 16; i += 256) dst[i] = src[i];
    }

    // ---- stage cos/sin of x on the 36x36 halo (zero outside: conv pads the
    //      cos/sin images with 0, not the angle) ----
    for (int i = tid; i < H2 * H2; i += 256) {
        const int yy = i / H2, xx = i % H2;
        const int gy = ty + yy - 2, gx = tx + xx - 2;
        float c = 0.0f, s = 0.0f;
        if (gy >= 0 && gy < LSZ && gx >= 0 && gx < LSZ) {
            __sincosf(xs[gy * LSZ + gx], &s, &c);
        }
        cs[yy][xx] = c; sn[yy][xx] = s;
    }
    __syncthreads();

    // ---- layer 1 on the 34x34 halo: (u1,v1) = rotate(b1) . normalize(conv) ----
    for (int i = tid; i < H1 * H1; i += 256) {
        const int hy = i / H1, hx = i % H1;
        const int gy = ty + hy - 1, gx = tx + hx - 1;
        _Float16* du = &U1[i * 16];
        _Float16* dv = &V1[i * 16];
        if (gy < 0 || gy >= LSZ || gx < 0 || gx >= LSZ) {
            #pragma unroll
            for (int c = 0; c < 16; ++c) { du[c] = (_Float16)0.0f; dv[c] = (_Float16)0.0f; }
        } else {
            float re[16], im[16];
            #pragma unroll
            for (int c = 0; c < 16; ++c) { re[c] = 0.0f; im[c] = 0.0f; }
            #pragma unroll
            for (int k = 0; k < 9; ++k) {
                const int ky = k / 3, kx = k % 3;
                const float cr = cs[hy + ky][hx + kx];
                const float ci = sn[hy + ky][hx + kx];
                #pragma unroll
                for (int c = 0; c < 16; ++c) {
                    const float w = w1s[c][k];
                    re[c] = fmaf(w, cr, re[c]);
                    im[c] = fmaf(w, ci, im[c]);
                }
            }
            #pragma unroll
            for (int c = 0; c < 16; ++c) {
                const float n2 = re[c] * re[c] + im[c] * im[c];
                const float rn = (n2 > 0.0f) ? rsqrtf(n2) : 0.0f;
                const float u  = (n2 > 0.0f) ? (re[c] * cb1s[c] - im[c] * sb1s[c]) * rn : cb1s[c];
                const float v  = (n2 > 0.0f) ? (im[c] * cb1s[c] + re[c] * sb1s[c]) * rn : sb1s[c];
                du[c] = (_Float16)u; dv[c] = (_Float16)v;
            }
        }
    }

    // conv2 A fragments -> registers (5 tap pairs, 40 VGPRs)
    v16h a2[5];
    #pragma unroll
    for (int tp = 0; tp < 5; ++tp)
        a2[tp] = *(const v16h*)(afrag2 + (tp * 32 + lane) * 16);

    __syncthreads();

    float pool[8];
    #pragma unroll
    for (int o = 0; o < 8; ++o) pool[o] = 0.0f;

    // ---- 64 row strips: 8 waves x 8 strips; uniform control flow for WMMA ----
    for (int si = 0; si < 8; ++si) {
        const int s   = wave * 8 + si;
        const int row = s >> 1;             // 0..31
        const int xh  = (s & 1) << 4;       // left/right 16-pixel half

        // conv2 as GEMM: D[16 o x 16 p] += A2[o,K] * B[K,p], K = (tap,c) pairs
        v8f accR = {}; v8f accI = {};
        #pragma unroll
        for (int tp = 0; tp < 5; ++tp) {
            const int t0  = 2 * tp;
            const int t1  = (2 * tp + 1 < 9) ? (2 * tp + 1) : t0; // dup tap, zeroed in A
            const int tap = hi ? t1 : t0;
            const int yy  = row + tap / 3;
            const int xx  = xh + p + tap % 3;
            const v16h bu = *(const v16h*)&U1[(yy * H1 + xx) * 16];
            const v16h bv = *(const v16h*)&V1[(yy * H1 + xx) * 16];
            accR = __builtin_amdgcn_wmma_f32_16x16x32_f16(false, a2[tp], false, bu,
                                                          (short)0, accR, false, false);
            accI = __builtin_amdgcn_wmma_f32_16x16x32_f16(false, a2[tp], false, bv,
                                                          (short)0, accI, false, false);
        }

        // normalize + rotate by b2: lane holds pixel p, channels {0..7}+8*hi
        float u2f[8], v2f[8];
        #pragma unroll
        for (int j = 0; j < 8; ++j) {
            const int o = j + hi * 8;
            const float rv = accR[j], iv = accI[j];
            const float n2 = rv * rv + iv * iv;
            const float rn = (n2 > 0.0f) ? rsqrtf(n2) : 0.0f;
            u2f[j] = (n2 > 0.0f) ? (rv * cb2s[o] - iv * sb2s[o]) * rn : cb2s[o];
            v2f[j] = (n2 > 0.0f) ? (iv * cb2s[o] + rv * sb2s[o]) * rn : sb2s[o];
        }

        // stage (u2,v2) in per-wave LDS so the packed B fragment is one b128x2
        {
            v8h uh, vh;
            #pragma unroll
            for (int j = 0; j < 8; ++j) { uh[j] = (_Float16)u2f[j]; vh[j] = (_Float16)v2f[j]; }
            *(v8h*)&SU[wave][p][hi * 8] = uh;
            *(v8h*)&SV[wave][p][hi * 8] = vh;
        }
        // same-wave LDS store->load: hardware keeps DS ops in order; fence the
        // compiler so it cannot sink the stores past the reload.
        asm volatile("" ::: "memory");

        // packed bilinear B fragment: K<16 -> u2[c], K>=16 -> v2[c]
        const _Float16* bsrc = hi ? &SV[wave][p][0] : &SU[wave][p][0];
        const v16h buv = *(const v16h*)bsrc;

        #pragma unroll
        for (int o = 0; o < 8; ++o) {
            const v16h au = *(const v16h*)&ABs[((o * 2 + 0) * 32 + lane) * 16];
            const v16h av = *(const v16h*)&ABs[((o * 2 + 1) * 32 + lane) * 16];
            v8f Y = {}; v8f Z = {};
            Y = __builtin_amdgcn_wmma_f32_16x16x32_f16(false, au, false, buv,
                                                       (short)0, Y, false, false);
            Z = __builtin_amdgcn_wmma_f32_16x16x32_f16(false, av, false, buv,
                                                       (short)0, Z, false, false);
            // quadratic forms: re = u.(W_o u) - v.(W_o v); im = u.(W_o v) + v.(W_o u)
            float uY = 0.f, vY = 0.f, uZ = 0.f, vZ = 0.f;
            #pragma unroll
            for (int j = 0; j < 8; ++j) {
                uY = fmaf(u2f[j], Y[j], uY);  vY = fmaf(v2f[j], Y[j], vY);
                uZ = fmaf(u2f[j], Z[j], uZ);  vZ = fmaf(v2f[j], Z[j], vZ);
            }
            uY += __shfl_xor(uY, 16, 32);  vY += __shfl_xor(vY, 16, 32);
            uZ += __shfl_xor(uZ, 16, 32);  vZ += __shfl_xor(vZ, 16, 32);
            const float re = uY - vZ;
            const float im = uZ + vY;
            float th = atan2f(im, re) + bb3s[o];
            th -= 6.28318530718f * rintf(th * 0.15915494309f);   // _wrap
            pool[o] += hi ? 0.0f : th;   // count each pixel once (select, no EXEC change)
        }
    }

    // ---- pooled reduction: wave shuffles -> LDS -> one atomic per (b,o) ----
    #pragma unroll
    for (int o = 0; o < 8; ++o) {
        float v = pool[o];
        v += __shfl_xor(v, 16, 32);
        v += __shfl_xor(v, 8, 32);
        v += __shfl_xor(v, 4, 32);
        v += __shfl_xor(v, 2, 32);
        v += __shfl_xor(v, 1, 32);
        if (lane == 0) red[wave][o] = v;
    }
    __syncthreads();
    if (tid < 8) {
        float s = 0.0f;
        #pragma unroll
        for (int w = 0; w < 8; ++w) s += red[w][tid];
        atomicAdd(&pooled[(slice >> 1) * 8 + tid], s);
    }
}

// ---------------------------------------------------------------------------
// Kernel 2: mean + fc1(relu) + fc2. One block per batch element, 64 threads.
// ---------------------------------------------------------------------------
__global__ void lcnn_head_kernel(const float* __restrict__ pooled,
                                 const float* __restrict__ fc1w,  // (64,8)
                                 const float* __restrict__ fc1b,  // (64)
                                 const float* __restrict__ fc2w,  // (1,64)
                                 const float* __restrict__ fc2b,  // (1)
                                 float* __restrict__ out) {
    const int b = blockIdx.x;
    const int t = threadIdx.x;       // 64 threads = 2 waves
    const float inv = 1.0f / (2.0f * 256.0f * 256.0f);
    float acc = fc1b[t];
    #pragma unroll
    for (int c = 0; c < 8; ++c)
        acc = fmaf(pooled[b * 8 + c] * inv, fc1w[t * 8 + c], acc);
    float z = fmaxf(acc, 0.0f) * fc2w[t];
    z += __shfl_xor(z, 16, 32);
    z += __shfl_xor(z, 8, 32);
    z += __shfl_xor(z, 4, 32);
    z += __shfl_xor(z, 2, 32);
    z += __shfl_xor(z, 1, 32);
    __shared__ float s2[2];
    if ((t & 31) == 0) s2[t >> 5] = z;
    __syncthreads();
    if (t == 0) out[b] = s2[0] + s2[1] + fc2b[0];
}

// ---------------------------------------------------------------------------
extern "C" void kernel_launch(void* const* d_in, const int* in_sizes, int n_in,
                              void* d_out, int out_size, void* d_ws, size_t ws_size,
                              hipStream_t stream) {
    (void)in_sizes; (void)n_in; (void)out_size; (void)ws_size;
    const float* x    = (const float*)d_in[0];
    const float* w1   = (const float*)d_in[1];
    const float* b1   = (const float*)d_in[2];
    const float* w2   = (const float*)d_in[3];
    const float* b2   = (const float*)d_in[4];
    const float* wb   = (const float*)d_in[5];
    const float* b3   = (const float*)d_in[6];
    const float* fc1w = (const float*)d_in[7];
    const float* fc1b = (const float*)d_in[8];
    const float* fc2w = (const float*)d_in[9];
    const float* fc2b = (const float*)d_in[10];
    float* out = (float*)d_out;

    char* ws = (char*)d_ws;
    float*    pooled = (float*)(ws + WS_POOLED_OFF);
    _Float16* afrag2 = (_Float16*)(ws + WS_AFRAG2_OFF);
    _Float16* afragb = (_Float16*)(ws + WS_AFRAGB_OFF);

    lcnn_setup_kernel<<<1, 256, 0, stream>>>(w2, wb, pooled, afrag2, afragb);
    lcnn_fused_kernel<<<64 * 64, 256, 0, stream>>>(x, w1, b1, b2, b3,
                                                   afrag2, afragb, pooled);
    lcnn_head_kernel<<<32, 64, 0, stream>>>(pooled, fc1w, fc1b, fc2w, fc2b, out);
}
